// PointConvs_67585605370026
// MI455X (gfx1250) — matrix operand order repaired and checked by reading, not compiled, not run on hardware
//
#include <hip/hip_runtime.h>

typedef __attribute__((ext_vector_type(16))) __bf16 v16bf;
typedef __attribute__((ext_vector_type(8)))  float  v8f;

#define NPTS   262144
#define LOCLEN 60
#define HID    32
#define NW     8
#define TILE_M 16
#define TILES  2                    // M-tiles per wave: B-fragment reuse x2
#define ROWS   (TILES * TILE_M)     // 32 points per wave
#define WAVES  8
#define BLK    (WAVES * 32)
#define NFRAG  29   // 4 (W0) + 8 (W1..W4) + 1 (W5) + 16 (bank)

static_assert(NPTS % (ROWS * WAVES) == 0, "grid tiling");

union FragB { v16bf v; __bf16 b[16]; unsigned int u[8]; };
union AccU  { v8f v; float f[8]; };
union PairU { __bf16 b[2]; unsigned int u; unsigned short s[2]; };

// native f32 -> bf16 (v_cvt_pk_bf16_f32 on gfx1250)
__device__ __forceinline__ __bf16 tobf(float x) { return (__bf16)x; }

// A-fragment K offset for vgpr-pair v (0..7), lane-half h (0/1)
__device__ __forceinline__ int kaOff(int v, int h) {
    return ((v >> 2) << 4) + (h << 3) + ((v & 3) << 1);
}
// LeakyReLU(0.01): for slope<1, identical to max(x, 0.01x)  -> v_mul + v_max
__device__ __forceinline__ float lrelu(float x) { return fmaxf(x, 0.01f * x); }

#define WMMA_BF16(A, B, C) \
    __builtin_amdgcn_wmma_f32_16x16x32_bf16(false, (A), false, (B), (short)0, (C), false, false)

__global__ __launch_bounds__(BLK) void pointconv_fused(
    const float* __restrict__ rel,  const float* __restrict__ feat,
    const float* __restrict__ w0,   const float* __restrict__ b0,
    const float* __restrict__ w1,   const float* __restrict__ b1,
    const float* __restrict__ w2,   const float* __restrict__ b2,
    const float* __restrict__ w3,   const float* __restrict__ b3,
    const float* __restrict__ w4,   const float* __restrict__ b4,
    const float* __restrict__ w5,   const float* __restrict__ b5,
    const float* __restrict__ bank, float* __restrict__ out)
{
    __shared__ unsigned int   s_frag[NFRAG * 32 * 8];     // B fragments, pre-swizzled (29.0 KB)
    __shared__ float          s_bias[6 * 32];             // biases, layer5 padded
    __shared__ unsigned short s_act [WAVES][ROWS * HID];  // per-wave activation staging (bf16, 2KB)
    __shared__ float          s_coef[WAVES][ROWS * NW];   // per-wave logits/softmax staging (1KB)

    const int tid = threadIdx.x;

    // ---------------- init: swizzle all weights into WMMA B-fragment layout ----------------
    const float* wl[4] = { w1, w2, w3, w4 };
    for (int u = tid; u < NFRAG * 32; u += BLK) {
        const int f = u >> 5, lane = u & 31;
        const int h = (lane >> 4) & 1, n = lane & 15;
        unsigned int* dst = &s_frag[(unsigned)u * 8u];
#pragma unroll
        for (int v = 0; v < 8; ++v) {
            const int k = kaOff(v, h);           // B layout matches A: K = 16*(v>>2) + 8h + 2*(v&3)
            float e0, e1;
            if (f < 4) {                          // W0 [60x32], K padded to 64
                const int kc = f >> 1, nh = f & 1;
                const int g0 = kc * 32 + k, g1 = g0 + 1;
                e0 = (g0 < LOCLEN) ? w0[g0 * HID + nh * 16 + n] : 0.f;
                e1 = (g1 < LOCLEN) ? w0[g1 * HID + nh * 16 + n] : 0.f;
            } else if (f < 12) {                  // W1..W4 [32x32]
                const int L = (f - 4) >> 1, nh = (f - 4) & 1;
                const float* w = wl[L];
                e0 = w[k * HID + nh * 16 + n];
                e1 = w[(k + 1) * HID + nh * 16 + n];
            } else if (f == 12) {                 // W5 [32x8], cols padded to 16
                e0 = (n < NW) ? w5[k * NW + n] : 0.f;
                e1 = (n < NW) ? w5[(k + 1) * NW + n] : 0.f;
            } else {                              // bank_j viewed as [32x32] row-major (K-major)
                const int j = (f - 13) >> 1, nh = (f - 13) & 1;
                e0 = bank[j * (HID * HID) + k * HID + nh * 16 + n];
                e1 = bank[j * (HID * HID) + (k + 1) * HID + nh * 16 + n];
            }
            PairU p; p.b[0] = tobf(e0); p.b[1] = tobf(e1);
            dst[v] = p.u;
        }
    }
    const float* bl[6] = { b0, b1, b2, b3, b4, b5 };
    for (int u = tid; u < 6 * 32; u += BLK) {
        const int L = u >> 5, o = u & 31;
        s_bias[u] = (L < 5) ? bl[L][o] : ((o < NW) ? bl[5][o] : 0.f);
    }
    __syncthreads();

    // ---------------- per-wave: 2 tiles of 16 points ----------------
    const int wv = tid >> 5, lane = tid & 31;
    const int h = (lane >> 4) & 1, nn = lane & 15;
    const int m0 = (blockIdx.x * WAVES + wv) * ROWS;
    unsigned short* act  = &s_act[wv][0];
    float*          coef = &s_coef[wv][0];

    auto loadB = [&](int f) -> v16bf {
        FragB b;
        const uint4* p = (const uint4*)&s_frag[((unsigned)(f * 32 + lane)) * 8u];
        ((uint4*)b.u)[0] = p[0];                     // ds_load_b128
        ((uint4*)b.u)[1] = p[1];                     // ds_load_b128
        return b.v;
    };
    auto loadActA = [&](int t) -> v16bf {            // activation (bf16 in LDS) -> A frag, tile t
        FragB A;
#pragma unroll
        for (int v = 0; v < 8; ++v) {
            const int k = kaOff(v, h);
            A.u[v] = *(const unsigned int*)&act[(t * TILE_M + nn) * HID + k];
        }
        return A.v;
    };
    auto actStore = [&](v8f C0, v8f C1, int L, int t) {  // bias + LeakyReLU + stage as bf16
        AccU u0, u1; u0.v = C0; u1.v = C1;
        const float bia0 = s_bias[L * 32 + nn];
        const float bia1 = s_bias[L * 32 + 16 + nn];
#pragma unroll
        for (int i = 0; i < 8; ++i) {
            const float x0 = lrelu(u0.f[i] + bia0);
            const float x1 = lrelu(u1.f[i] + bia1);
            const int row = t * TILE_M + h * 8 + i;
            PairU p0; p0.b[0] = tobf(x0);
            PairU p1; p1.b[0] = tobf(x1);
            act[row * HID + nn]      = p0.s[0];
            act[row * HID + 16 + nn] = p1.s[0];
        }
    };

    // ----- layer 0: [16x64(pad)] @ [64x32] per tile, B loaded once -----
    v16bf A0[TILES], A1[TILES];
#pragma unroll
    for (int t = 0; t < TILES; ++t) {
        FragB F0, F1;
        const float* rp = rel + (size_t)(m0 + t * TILE_M + nn) * LOCLEN;
#pragma unroll
        for (int v = 0; v < 8; ++v) {
            const int k = kaOff(v, h);
            F0.b[2 * v]     = tobf(rp[k]);
            F0.b[2 * v + 1] = tobf(rp[k + 1]);
            const int g = 32 + k;
            F1.b[2 * v]     = tobf((g     < LOCLEN) ? rp[g]     : 0.f);
            F1.b[2 * v + 1] = tobf((g + 1 < LOCLEN) ? rp[g + 1] : 0.f);
        }
        A0[t] = F0.v; A1[t] = F1.v;
    }
    {
        const v16bf B0 = loadB(0), B1 = loadB(1), B2 = loadB(2), B3 = loadB(3);
#pragma unroll
        for (int t = 0; t < TILES; ++t) {
            v8f c0 = {}, c1 = {};
            c0 = WMMA_BF16(A0[t], B0, c0);
            c0 = WMMA_BF16(A1[t], B2, c0);
            c1 = WMMA_BF16(A0[t], B1, c1);
            c1 = WMMA_BF16(A1[t], B3, c1);
            actStore(c0, c1, 0, t);
        }
    }

    // ----- layers 1..4: [16x32] @ [32x32], B loaded once per layer -----
#pragma unroll
    for (int L = 1; L <= 4; ++L) {
        v16bf a[TILES];
#pragma unroll
        for (int t = 0; t < TILES; ++t) a[t] = loadActA(t);
        const v16bf Bl0 = loadB(4 + (L - 1) * 2 + 0);
        const v16bf Bl1 = loadB(4 + (L - 1) * 2 + 1);
#pragma unroll
        for (int t = 0; t < TILES; ++t) {
            v8f d0 = {}, d1 = {};
            d0 = WMMA_BF16(a[t], Bl0, d0);
            d1 = WMMA_BF16(a[t], Bl1, d1);
            actStore(d0, d1, L, t);
        }
    }

    // ----- layer 5: logits [16x8] per tile -----
    {
        const v16bf B5 = loadB(12);
#pragma unroll
        for (int t = 0; t < TILES; ++t) {
            v16bf a = loadActA(t);
            v8f l5 = {};
            l5 = WMMA_BF16(a, B5, l5);
            AccU u; u.v = l5;
            if (nn < NW) {
                const float bia = s_bias[5 * 32 + nn];
#pragma unroll
                for (int i = 0; i < 8; ++i)
                    coef[(t * TILE_M + h * 8 + i) * NW + nn] = u.f[i] + bia;
            }
        }
    }
    // ----- softmax: all 32 lanes, one row each (in-order DS within wave) -----
    {
        float v[NW];
#pragma unroll
        for (int j = 0; j < NW; ++j) v[j] = coef[lane * NW + j];
        float m = v[0];
#pragma unroll
        for (int j = 1; j < NW; ++j) m = fmaxf(m, v[j]);
        float s = 0.f;
#pragma unroll
        for (int j = 0; j < NW; ++j) { v[j] = __expf(v[j] - m); s += v[j]; }
        const float inv = 1.f / s;
#pragma unroll
        for (int j = 0; j < NW; ++j) coef[lane * NW + j] = v[j] * inv;
    }

    // ----- final: out = [coeff (x) feat] (16x256) @ stacked bank (256x32) per tile -----
    __bf16 cb[TILES][NW];
#pragma unroll
    for (int t = 0; t < TILES; ++t)
#pragma unroll
        for (int j = 0; j < NW; ++j)
            cb[t][j] = tobf(coef[(t * TILE_M + nn) * NW + j]);

    FragB FB[TILES];                                 // feat A fragments in bf16
#pragma unroll
    for (int t = 0; t < TILES; ++t) {
        const float* fp = feat + (size_t)(m0 + t * TILE_M + nn) * HID;
#pragma unroll
        for (int v = 0; v < 8; ++v) {
            const int k = kaOff(v, h);
            FB[t].b[2 * v]     = tobf(fp[k]);
            FB[t].b[2 * v + 1] = tobf(fp[k + 1]);
        }
    }
    v8f o0[TILES] = {}, o1[TILES] = {};
#pragma unroll
    for (int j = 0; j < NW; ++j) {
        const v16bf Bj0 = loadB(13 + j * 2 + 0);
        const v16bf Bj1 = loadB(13 + j * 2 + 1);
#pragma unroll
        for (int t = 0; t < TILES; ++t) {
            FragB A;
#pragma unroll
            for (int k = 0; k < 16; ++k) A.b[k] = FB[t].b[k] * cb[t][j];  // v_pk_mul_bf16
            o0[t] = WMMA_BF16(A.v, Bj0, o0[t]);
            o1[t] = WMMA_BF16(A.v, Bj1, o1[t]);
        }
    }

    // ----- store [32x32] f32 -----
#pragma unroll
    for (int t = 0; t < TILES; ++t) {
        AccU U0, U1; U0.v = o0[t]; U1.v = o1[t];
        float* op = out + (size_t)(m0 + t * TILE_M) * HID;
#pragma unroll
        for (int i = 0; i < 8; ++i) {
            const int row = h * 8 + i;
            op[row * HID + nn]      = U0.f[i];
            op[row * HID + 16 + nn] = U1.f[i];
        }
    }
}

extern "C" void kernel_launch(void* const* d_in, const int* in_sizes, int n_in,
                              void* d_out, int out_size, void* d_ws, size_t ws_size,
                              hipStream_t stream) {
    (void)in_sizes; (void)n_in; (void)out_size; (void)d_ws; (void)ws_size;
    const float* rel  = (const float*)d_in[0];
    const float* feat = (const float*)d_in[1];
    const float* w0 = (const float*)d_in[2];  const float* b0 = (const float*)d_in[3];
    const float* w1 = (const float*)d_in[4];  const float* b1 = (const float*)d_in[5];
    const float* w2 = (const float*)d_in[6];  const float* b2 = (const float*)d_in[7];
    const float* w3 = (const float*)d_in[8];  const float* b3 = (const float*)d_in[9];
    const float* w4 = (const float*)d_in[10]; const float* b4 = (const float*)d_in[11];
    const float* w5 = (const float*)d_in[12]; const float* b5 = (const float*)d_in[13];
    const float* bank = (const float*)d_in[14];

    const int blocks = NPTS / (ROWS * WAVES);   // 1024 blocks; 8 waves x 32 points each
    pointconv_fused<<<blocks, BLK, 0, stream>>>(rel, feat, w0, b0, w1, b1, w2, b2,
                                                w3, b3, w4, b4, w5, b5, bank,
                                                (float*)d_out);
}
